// CompoundHeadAttention_49512382988483
// MI455X (gfx1250) — matrix operand order repaired and checked by reading, not compiled, not run on hardware
//
#include <hip/hip_runtime.h>
#include <hip/hip_bf16.h>

// ---------------------------------------------------------------------------
// CompoundHeadAttention for MI455X (gfx1250): all matmuls via
// v_wmma_f32_16x16x32_bf16 (wave32 WMMA), softmax in f32.
//   - bf16 GEMMs with 64x64 wave tiles (16 WMMA per K-step per wave)
//   - causal flash attention, 32 query rows per wave32:
//       * softmax temperature AND log2(e) folded into Q fragments ->
//         exp2-domain online softmax (v_exp_f32 with no extra multiply)
//       * causal mask applied only on the single diagonal tile
//       * softmax row-sums via WMMA against an all-ones matrix
//       * row-max butterfly via v_permlane16_b32 (VALU, no DS waits)
//   - LDS-tiled weight transposes, vectorized dtype conversion
// ---------------------------------------------------------------------------

typedef __bf16 bf16;
typedef __attribute__((ext_vector_type(8)))  __bf16 bf16x8;
typedef __attribute__((ext_vector_type(16))) __bf16 bf16x16;
typedef __attribute__((ext_vector_type(8)))  float  f32x8;
typedef __attribute__((ext_vector_type(4)))  float  f32x4;

#define WMMA_BF16(A, B, C) \
  __builtin_amdgcn_wmma_f32_16x16x32_bf16(false, (A), false, (B), (short)0, (C), false, false)

static constexpr int N_TOK = 2048;   // sequence length
static constexpr int E     = 2048;   // embed dim
static constexpr int HKV   = 8;      // kv heads
static constexpr int GRP   = 4;      // query groups per kv head
static constexpr int DH    = 64;     // head dim
static constexpr int KV_E  = 512;    // HKV * DH
static constexpr int EHK   = 256;    // GRP * DH

// softmax(scores / D^-0.5) == softmax(8*scores); run exp in base-2:
// fold 8 * log2(e) into Q so P = exp2(sc - mn) directly.
static constexpr float QSCALE = 8.0f * 1.44269504088896340736f;

// Device base-2 exponential (hardware v_exp_f32 is base-2 natively).
static __device__ __forceinline__ float fast_exp2(float x) {
#if __has_builtin(__builtin_amdgcn_exp2f)
  return __builtin_amdgcn_exp2f(x);
#else
  return exp2f(x);
#endif
}

// Merge two contiguous 16-byte bf16 loads into one 16-element WMMA fragment.
static __device__ __forceinline__ bf16x16 frag16(const bf16* p0, const bf16* p1) {
  bf16x8 a = *(const bf16x8*)p0;
  bf16x8 b = *(const bf16x8*)p1;
  return __builtin_shufflevector(a, b, 0,1,2,3,4,5,6,7,8,9,10,11,12,13,14,15);
}

// Same, but pre-scaled by QSCALE (softmax temperature + exp2 conversion).
static __device__ __forceinline__ bf16x16 frag16_qs(const bf16* p0, const bf16* p1) {
  bf16x16 r = frag16(p0, p1);
#pragma unroll
  for (int e = 0; e < 16; ++e) r[e] = (bf16)((float)r[e] * QSCALE);
  return r;
}

// Max across each 16-lane group, result replicated to all 16 lanes.
// v_permlane16_b32 is a VALU 16-lane permute (two nibble-select immediates,
// lanes 0-7 in s1, lanes 8-15 in s2) -> no LDS round trip, no dscnt waits.
static __device__ __forceinline__ float rowmax16(float v) {
#if __has_builtin(__builtin_amdgcn_permlane16)
  unsigned u, p;
  u = __float_as_uint(v);   // xor 1
  p = __builtin_amdgcn_permlane16(u, u, 0x67452301u, 0xEFCDAB89u, false, false);
  v = fmaxf(v, __uint_as_float(p));
  u = __float_as_uint(v);   // xor 2
  p = __builtin_amdgcn_permlane16(u, u, 0x54761032u, 0xDCFE98BAu, false, false);
  v = fmaxf(v, __uint_as_float(p));
  u = __float_as_uint(v);   // xor 4
  p = __builtin_amdgcn_permlane16(u, u, 0x32107654u, 0xBA98FEDCu, false, false);
  v = fmaxf(v, __uint_as_float(p));
  u = __float_as_uint(v);   // xor 8
  p = __builtin_amdgcn_permlane16(u, u, 0xFEDCBA98u, 0x76543210u, false, false);
  v = fmaxf(v, __uint_as_float(p));
#else
  v = fmaxf(v, __shfl_xor(v, 1, 32));
  v = fmaxf(v, __shfl_xor(v, 2, 32));
  v = fmaxf(v, __shfl_xor(v, 4, 32));
  v = fmaxf(v, __shfl_xor(v, 8, 32));
#endif
  return v;
}

// ---------------------------------------------------------------------------
// f32 -> bf16 conversion, 8 elements per lane (b128 in, b128 out)
// ---------------------------------------------------------------------------
__global__ void cvt_bf16_kernel(const float* __restrict__ src, bf16* __restrict__ dst, int n) {
  int i = (blockIdx.x * blockDim.x + threadIdx.x) * 8;
  const int stride = gridDim.x * blockDim.x * 8;
  for (; i < n; i += stride) {
    f32x4 f0 = *(const f32x4*)(src + i);
    f32x4 f1 = *(const f32x4*)(src + i + 4);
    bf16x8 o;
#pragma unroll
    for (int e = 0; e < 4; ++e) { o[e] = (bf16)f0[e]; o[e + 4] = (bf16)f1[e]; }
    *(bf16x8*)(dst + i) = o;
  }
}

// ---------------------------------------------------------------------------
// f32 [rows x cols] -> bf16 [cols x rows], LDS-tiled (coalesced both ways),
// batched via blockIdx.z. rows/cols are multiples of 16.
// ---------------------------------------------------------------------------
__global__ void transpose_cvt_kernel(const float* __restrict__ src, bf16* __restrict__ dst,
                                     int rows, int cols, long sBatch, long dBatch) {
  __shared__ bf16 tile[16][17];
  const float* s = src + (long)blockIdx.z * sBatch;
  bf16*        d = dst + (long)blockIdx.z * dBatch;
  const int k0 = blockIdx.y * 16;   // src row base
  const int n0 = blockIdx.x * 16;   // src col base
  const int tx = threadIdx.x, ty = threadIdx.y;
  tile[ty][tx] = (bf16)s[(long)(k0 + ty) * cols + n0 + tx];   // coalesced read
  __syncthreads();
  d[(long)(n0 + ty) * rows + k0 + tx] = tile[tx][ty];         // coalesced write
}

// ---------------------------------------------------------------------------
// bf16 GEMM, f32 accumulate:
//   C[M x N] = A[M x K] (row-major) * Bt[N x K] (pre-transposed) + bias[N]
// Block = 256 threads = 8 wave32s arranged 2(M) x 4(N); wave tile 64 x 64
// (4x4 fragments -> 16 WMMA per 32-wide K step, 16B loaded per WMMA);
// block tile 128 x 256. Each B fragment is consumed by 4 WMMAs right after
// load, so only ~1 B fragment is live at a time (acc dominates registers).
// OUT_MODE: 0 -> bf16 row-major, 1 -> bf16 transposed (C^T), 2 -> f32 row-major
// ---------------------------------------------------------------------------
template <int OUT_MODE>
__launch_bounds__(256)
__global__ void gemm_bf16_wmma(const bf16* __restrict__ A, int lda, long aBatch,
                               const bf16* __restrict__ Bt, int ldb, long bBatch,
                               const float* __restrict__ bias, long biasBatch,
                               void* __restrict__ Cv, int ldc, long cBatch, int K) {
  A  += (long)blockIdx.z * aBatch;
  Bt += (long)blockIdx.z * bBatch;
  bias += (long)blockIdx.z * biasBatch;

  const int lane = threadIdx.x & 31;
  const int wid  = threadIdx.x >> 5;
  const int wy   = wid & 1;            // 2 waves along M
  const int wx   = wid >> 1;           // 4 waves along N
  const int mBase = blockIdx.y * 128 + wy * 64;
  const int nBase = blockIdx.x * 256 + wx * 64;
  const int half = lane >> 4;          // 0 | 1
  const int l16  = lane & 15;

  f32x8 acc[4][4];
#pragma unroll
  for (int i = 0; i < 4; ++i)
#pragma unroll
    for (int j = 0; j < 4; ++j) acc[i][j] = {};

  for (int kb = 0; kb < K; kb += 32) {
    bf16x16 a[4];
#pragma unroll
    for (int i = 0; i < 4; ++i) {
      const bf16* ap = A + (long)(mBase + i * 16 + l16) * lda + kb + half * 8;
      a[i] = frag16(ap, ap + 16);
    }
#pragma unroll
    for (int j = 0; j < 4; ++j) {
      const bf16* bp = Bt + (long)(nBase + j * 16 + l16) * ldb + kb + half * 8;
      bf16x16 b = frag16(bp, bp + 16);
#pragma unroll
      for (int i = 0; i < 4; ++i) acc[i][j] = WMMA_BF16(a[i], b, acc[i][j]);
    }
  }

  // Epilogue. C/D layout: within 16x16, row = r + 8*half, col = l16.
#pragma unroll
  for (int i = 0; i < 4; ++i) {
#pragma unroll
    for (int j = 0; j < 4; ++j) {
      const int colg = nBase + j * 16 + l16;
      const float bb = bias[colg];
#pragma unroll
      for (int r = 0; r < 8; ++r) {
        const int rowg = mBase + i * 16 + r + half * 8;
        const float v = acc[i][j][r] + bb;
        if (OUT_MODE == 0) {
          ((bf16*)Cv + (long)blockIdx.z * cBatch)[(long)rowg * ldc + colg] = (bf16)v;
        } else if (OUT_MODE == 1) {
          ((bf16*)Cv + (long)blockIdx.z * cBatch)[(long)colg * ldc + rowg] = (bf16)v;
        } else {
          ((float*)Cv + (long)blockIdx.z * cBatch)[(long)rowg * ldc + colg] = v;
        }
      }
    }
  }
}

// ---------------------------------------------------------------------------
// Causal flash attention, one wave32 per block, 32 query rows per block.
//   Qg  : [N_TOK][E]    bf16, query head (h,g) at cols h*256 + g*64 .. +63
//   Kp  : [N_TOK][KV_E] bf16, kv head h at cols h*64 .. +63
//   Vt  : [KV_E][N_TOK] bf16 (V transposed)
//   Aout: [N_TOK][E]    bf16, output head (h,g) at cols (h*G+g)*64 .. +63
// QSCALE pre-folded into Q; softmax runs in exp2 domain. Row tiles and s
// tiles are both 32-aligned, so only the tile s0 == n0 needs the causal mask.
// ---------------------------------------------------------------------------
__launch_bounds__(32)
__global__ void flash_attn_kernel(const bf16* __restrict__ Qg, const bf16* __restrict__ Kp,
                                  const bf16* __restrict__ Vt, bf16* __restrict__ Aout) {
  __shared__ __align__(16) bf16 pShm[32 * 32];   // P tile restage (C-layout -> A-layout)

  const int lane = threadIdx.x & 31;
  const int half = lane >> 4;
  const int l16  = lane & 15;
  const int n0   = blockIdx.x * 32;              // query row tile (32 rows)
  const int qh   = blockIdx.y;                   // 0..31
  const int h    = qh >> 2;
  const int g    = qh & 3;

  // Constant all-ones B fragment: row-sum via WMMA (sum lands in every column)
  bf16x16 onesB;
#pragma unroll
  for (int e = 0; e < 16; ++e) onesB[e] = (bf16)1.0f;

  // Q fragments: 2 row-frags x 2 d-chunks, pre-scaled, kept in registers
  bf16x16 qa[2][2];
#pragma unroll
  for (int i = 0; i < 2; ++i) {
    const bf16* qrow = Qg + (long)(n0 + i * 16 + l16) * E + h * EHK + g * DH;
#pragma unroll
    for (int c = 0; c < 2; ++c) {
      const bf16* p = qrow + c * 32 + half * 8;
      qa[i][c] = frag16_qs(p, p + 16);
    }
  }

  float m[2][8];
  f32x8 lv[2];                                   // row sums, replicated per column
  f32x8 o[2][4];
#pragma unroll
  for (int i = 0; i < 2; ++i) {
#pragma unroll
    for (int r = 0; r < 8; ++r) m[i][r] = -3.0e38f;
    lv[i] = {};
#pragma unroll
    for (int jd = 0; jd < 4; ++jd) o[i][jd] = {};
  }

  const bf16* KpH = Kp + h * DH;
  const bf16* VtH = Vt + (long)(h * DH) * N_TOK;

  auto process_tile = [&](int s0, bool masked) {
    // ---- K fragments for this s-tile (shared by both row-frags) ----
    bf16x16 kb[2][2];
#pragma unroll
    for (int j = 0; j < 2; ++j)
#pragma unroll
      for (int c = 0; c < 2; ++c) {
        const bf16* kp = KpH + (long)(s0 + j * 16 + l16) * KV_E + c * 32 + half * 8;
        kb[j][c] = frag16(kp, kp + 16);
      }

    // ---- scores S[32 x 32] = (QSCALE*Q) . K^T over d=64 ----
    f32x8 sc[2][2];
#pragma unroll
    for (int i = 0; i < 2; ++i)
#pragma unroll
      for (int j = 0; j < 2; ++j) {
        sc[i][j] = {};
#pragma unroll
        for (int c = 0; c < 2; ++c) sc[i][j] = WMMA_BF16(qa[i][c], kb[j][c], sc[i][j]);
      }

    // ---- (diagonal tile only) causal mask ----
    if (masked) {
#pragma unroll
      for (int i = 0; i < 2; ++i)
#pragma unroll
        for (int j = 0; j < 2; ++j)
#pragma unroll
          for (int r = 0; r < 8; ++r) {
            const int row = n0 + i * 16 + r + half * 8;
            const int col = s0 + j * 16 + l16;
            if (col > row) sc[i][j][r] = -3.0e38f;
          }
    }

    // ---- online softmax per row-frag (exp2 domain) ----
#pragma unroll
    for (int i = 0; i < 2; ++i) {
      float rescale[8];
#pragma unroll
      for (int r = 0; r < 8; ++r) {
        const float vmax = rowmax16(fmaxf(sc[i][0][r], sc[i][1][r]));
        const float mn = fmaxf(m[i][r], vmax);
        rescale[r] = fast_exp2(m[i][r] - mn);
        m[i][r] = mn;
#pragma unroll
        for (int j = 0; j < 2; ++j) sc[i][j][r] = fast_exp2(sc[i][j][r] - mn);
      }
      // rescale running sum and output accumulators
#pragma unroll
      for (int r = 0; r < 8; ++r) {
        lv[i][r] *= rescale[r];
#pragma unroll
        for (int jd = 0; jd < 4; ++jd) o[i][jd][r] *= rescale[r];
      }
      // stage P (bf16) into LDS in C-layout
#pragma unroll
      for (int j = 0; j < 2; ++j)
#pragma unroll
        for (int r = 0; r < 8; ++r)
          pShm[(i * 16 + r + half * 8) * 32 + j * 16 + l16] = (bf16)sc[i][j][r];
    }

    // Single wave per block: DS ops are in-order within a wave -> RAW safe.
    bf16x16 pa[2];
#pragma unroll
    for (int i = 0; i < 2; ++i) {
      const int base = (i * 16 + l16) * 32 + half * 8;
      pa[i] = frag16(&pShm[base], &pShm[base + 16]);
      // l += rowsum(P): WMMA against all-ones, accumulated into rescaled lv
      lv[i] = WMMA_BF16(pa[i], onesB, lv[i]);
    }

    // ---- O += P . V  (K=32 over s; 4 output d-blocks; V frags reused) ----
#pragma unroll
    for (int jd = 0; jd < 4; ++jd) {
      const bf16* vp = VtH + (long)(jd * 16 + l16) * N_TOK + s0 + half * 8;
      bf16x16 vb = frag16(vp, vp + 16);
#pragma unroll
      for (int i = 0; i < 2; ++i) o[i][jd] = WMMA_BF16(pa[i], vb, o[i][jd]);
    }
  };

  for (int s0 = 0; s0 < n0; s0 += 32) process_tile(s0, false);  // unmasked body
  process_tile(n0, true);                                       // diagonal tile

  // ---- normalized output: embed col = (h*G + g)*64 + d ----
  const int obase = (h * GRP + g) * DH;
#pragma unroll
  for (int i = 0; i < 2; ++i) {
    float invl[8];
#pragma unroll
    for (int r = 0; r < 8; ++r) invl[r] = 1.0f / lv[i][r];
#pragma unroll
    for (int jd = 0; jd < 4; ++jd)
#pragma unroll
      for (int r = 0; r < 8; ++r) {
        const int row = n0 + i * 16 + r + half * 8;
        Aout[(long)row * E + obase + jd * 16 + l16] = (bf16)(o[i][jd][r] * invl[r]);
      }
  }
}

// ---------------------------------------------------------------------------
// Host orchestration
// ---------------------------------------------------------------------------
extern "C" void kernel_launch(void* const* d_in, const int* in_sizes, int n_in,
                              void* d_out, int out_size, void* d_ws, size_t ws_size,
                              hipStream_t stream) {
  (void)in_sizes; (void)n_in; (void)out_size; (void)ws_size;
  const float* q   = (const float*)d_in[0];
  const float* k   = (const float*)d_in[1];
  const float* v   = (const float*)d_in[2];
  const float* Wq  = (const float*)d_in[3];
  const float* bq  = (const float*)d_in[4];
  const float* Wk  = (const float*)d_in[5];
  const float* bk  = (const float*)d_in[6];
  const float* Wv  = (const float*)d_in[7];
  const float* bv  = (const float*)d_in[8];
  const float* WG  = (const float*)d_in[9];
  const float* bG  = (const float*)d_in[10];
  const float* Wfc = (const float*)d_in[11];
  const float* bfc = (const float*)d_in[12];
  float* out = (float*)d_out;

  // Workspace carve-up (256B aligned)
  char* ws = (char*)d_ws;
  size_t off = 0;
  auto carve = [&](size_t bytes) -> char* {
    char* p = ws + off;
    off += (bytes + 255) & ~(size_t)255;
    return p;
  };
  const size_t SZ_NE  = (size_t)N_TOK * E * 2;       // 8 MB
  const size_t SZ_NKV = (size_t)N_TOK * KV_E * 2;    // 2 MB
  bf16* qb   = (bf16*)carve(SZ_NE);
  bf16* kb   = (bf16*)carve(SZ_NE);
  bf16* vb   = (bf16*)carve(SZ_NE);
  bf16* Wqt  = (bf16*)carve(SZ_NKV);                 // [KV_E][E]
  bf16* Wkt  = (bf16*)carve(SZ_NKV);
  bf16* Wvt  = (bf16*)carve(SZ_NKV);
  bf16* WGt  = (bf16*)carve((size_t)HKV * EHK * DH * 2);  // [h][EHK][DH]
  bf16* Wfct = (bf16*)carve(SZ_NE);                  // [E][E] transposed
  bf16* Qp   = (bf16*)carve(SZ_NKV);                 // [N][KV_E]
  bf16* Kp   = (bf16*)carve(SZ_NKV);                 // [N][KV_E]
  bf16* Vt   = (bf16*)carve(SZ_NKV);                 // [KV_E][N]
  bf16* Qg   = (bf16*)carve(SZ_NE);                  // [N][E] (h*256 + g*64 + d)
  bf16* Aout = (bf16*)carve(SZ_NE);                  // [N][E] ((h*G+g)*64 + d)

  // 1) convert activations to bf16 (vectorized)
  const int ne = N_TOK * E;
  cvt_bf16_kernel<<<512, 256, 0, stream>>>(q, qb, ne);
  cvt_bf16_kernel<<<512, 256, 0, stream>>>(k, kb, ne);
  cvt_bf16_kernel<<<512, 256, 0, stream>>>(v, vb, ne);

  // 2) transpose + convert weights: Wt[n][k] = W[k][n]
  {
    dim3 b(16, 16);
    transpose_cvt_kernel<<<dim3(KV_E / 16, E / 16, 1), b, 0, stream>>>(Wq, Wqt, E, KV_E, 0, 0);
    transpose_cvt_kernel<<<dim3(KV_E / 16, E / 16, 1), b, 0, stream>>>(Wk, Wkt, E, KV_E, 0, 0);
    transpose_cvt_kernel<<<dim3(KV_E / 16, E / 16, 1), b, 0, stream>>>(Wv, Wvt, E, KV_E, 0, 0);
    transpose_cvt_kernel<<<dim3(EHK / 16, DH / 16, HKV), b, 0, stream>>>(
        WG, WGt, DH, EHK, (long)DH * EHK, (long)DH * EHK);
    transpose_cvt_kernel<<<dim3(E / 16, E / 16, 1), b, 0, stream>>>(Wfc, Wfct, E, E, 0, 0);
  }

  // 3) projections (M=2048, N=512, K=2048); V written transposed -> Vt[KV_E][N]
  {
    dim3 grid(KV_E / 256, N_TOK / 128, 1);
    gemm_bf16_wmma<0><<<grid, 256, 0, stream>>>(qb, E, 0, Wqt, E, 0, bq, 0, Qp, KV_E, 0, E);
    gemm_bf16_wmma<0><<<grid, 256, 0, stream>>>(kb, E, 0, Wkt, E, 0, bk, 0, Kp, KV_E, 0, E);
    gemm_bf16_wmma<1><<<grid, 256, 0, stream>>>(vb, E, 0, Wvt, E, 0, bv, 0, Vt, N_TOK, 0, E);
  }

  // 4) per-kv-head compound transform: Qg[:, h*256:] = Qp[:, h*64:] @ WG[h] + bG[h]
  {
    dim3 grid(EHK / 256, N_TOK / 128, HKV);
    gemm_bf16_wmma<0><<<grid, 256, 0, stream>>>(Qp, KV_E, (long)DH,
                                                WGt, DH, (long)EHK * DH,
                                                bG, (long)EHK,
                                                Qg, E, (long)EHK, DH);
  }

  // 5) causal flash attention: 32 heads x 64 row tiles, 1 wave32 per block
  flash_attn_kernel<<<dim3(N_TOK / 32, HKV * GRP), 32, 0, stream>>>(Qg, Kp, Vt, Aout);

  // 6) final projection -> f32 output
  {
    dim3 grid(E / 256, N_TOK / 128, 1);
    gemm_bf16_wmma<2><<<grid, 256, 0, stream>>>(Aout, E, 0, Wfct, E, 0, bfc, 0, out, E, 0, E);
  }
}
